// MultiHeadCrossAttention_40922448396561
// MI455X (gfx1250) — matrix operand order repaired and checked by reading, not compiled, not run on hardware
//
#include <hip/hip_runtime.h>

// ---------------------------------------------------------------------------
// MultiHeadCrossAttention on MI455X (gfx1250, wave32, WMMA bf16)
// B=8 N=8 Q=128 K=1024 D=512 H=8 HD=64
// Pipeline: q/k/v projections (bf16 WMMA GEMM; V written head-transposed) ->
//           fused flash attention (double-buffered async LDS staging,
//           WMMA qk^T + online softmax + WMMA p@v) -> output projection.
// CDNA5 paths: v_wmma_f32_16x16x32_bf16, global_load_async_to_lds_b128
//              (ASYNCcnt) double-buffered staging, global_prefetch_b8.
// ---------------------------------------------------------------------------

typedef __attribute__((ext_vector_type(16))) __bf16 v16bf;
typedef __attribute__((ext_vector_type(8)))  __bf16 v8bf;
typedef __attribute__((ext_vector_type(8)))  float  v8f;

constexpr int CB  = 8;
constexpr int CN  = 8;
constexpr int CQ  = 128;
constexpr int CK  = 1024;
constexpr int CD  = 512;
constexpr int CH  = 8;
constexpr int CHD = 64;
constexpr float CSCALE   = 0.125f;                  // 64^-0.5
constexpr float CINV2SIG = 1.0f / (2.0f * 0.1225f); // sigma = 0.35

__device__ __forceinline__ __bf16 f2bf(float f) {
  unsigned u = __builtin_bit_cast(unsigned, f);
  u += 0x7FFFu + ((u >> 16) & 1u);               // round-to-nearest-even
  unsigned short h = (unsigned short)(u >> 16);
  return __builtin_bit_cast(__bf16, h);
}

__device__ __forceinline__ v16bf cat8(v8bf lo, v8bf hi) {
  return __builtin_shufflevector(lo, hi, 0,1,2,3,4,5,6,7,8,9,10,11,12,13,14,15);
}

__device__ __forceinline__ v8f wmma_bf16(v16bf a, v16bf b, v8f c) {
  // D = A(16x32 bf16) * B(32x16 bf16) + C(16x16 f32)
  return __builtin_amdgcn_wmma_f32_16x16x32_bf16(false, a, false, b,
                                                 (short)0, c, false, false);
}

// Async copy of 64 contiguous bytes global -> LDS (4 x b128, ASYNCcnt).
// INST_OFFSET is added to BOTH the LDS and global addresses (ISA 08 §4.4),
// so one LDS-addr VGPR + one 64-bit global base covers all four beats.
__device__ __forceinline__ void async_copy64(const void* lds_dst,
                                             const void* g_src) {
  unsigned           l = (unsigned)(unsigned long long)(uintptr_t)lds_dst;
  unsigned long long g = (unsigned long long)(uintptr_t)g_src;
  asm volatile(
      "global_load_async_to_lds_b128 %0, %1, off\n\t"
      "global_load_async_to_lds_b128 %0, %1, off offset:16\n\t"
      "global_load_async_to_lds_b128 %0, %1, off offset:32\n\t"
      "global_load_async_to_lds_b128 %0, %1, off offset:48"
      :: "v"(l), "v"(g) : "memory");
}

__device__ __forceinline__ void wait_async0() {
  asm volatile("s_wait_asynccnt 0x0" ::: "memory");
}

// ---------------------------------------------------------------------------
// GEMM: Y[M,512] = X[M,512] @ W[512,512]^T + bias   (nn.Linear semantics)
// Block tile 128x128, 8 waves, each wave 32x64 (2x4 WMMA subtiles), K-step 32.
// OUT_TRANSV: write bf16 result head-transposed as [bn][h][hd][key] so the
// attention kernel can stage V^T by straight async DMA (no LDS transpose).
// ---------------------------------------------------------------------------
template<bool IN_BF16, bool OUT_BF16, bool OUT_TRANSV>
__global__ __launch_bounds__(256) void gemm_xwt_512(
    const void* __restrict__ Xv, const float* __restrict__ W,
    const float* __restrict__ bias, void* __restrict__ Yv)
{
  __shared__ __bf16 As[128][40];   // [m][k]  (pad 40 to stagger banks)
  __shared__ __bf16 Bs[128][40];   // [n][k]  (B[k][n] = W[n][k])

  const int tid  = threadIdx.x;
  const int lane = tid & 31;
  const int w    = tid >> 5;
  const int lh   = lane >> 4;      // half of wave (0/1)
  const int r    = lane & 15;

  const int m0 = (blockIdx.x >> 2) * 128;
  const int n0 = (blockIdx.x & 3) * 128;

  const int srow = tid >> 1;           // staging row 0..127
  const int scol = (tid & 1) * 16;     // staging col 0/16

  const int mBase = (w & 3) * 32;      // wave's M stripe
  const int nBase = (w >> 2) * 64;     // wave's N stripe

  v8f c[2][4] = {};

  const float*  Xf = (const float*)Xv;
  const __bf16* Xb = (const __bf16*)Xv;

  const size_t xrow = (size_t)(m0 + srow) * CD;
  const size_t wrow = (size_t)(n0 + srow) * CD;

  for (int k0 = 0; k0 < CD; k0 += 32) {
    __syncthreads();
    // prefetch next K-step's lines while we stage this one
    if (k0 + 32 < CD) {
      if constexpr (IN_BF16)
        __builtin_prefetch(Xb + xrow + k0 + 32 + scol, 0, 0);
      else
        __builtin_prefetch(Xf + xrow + k0 + 32 + scol, 0, 0);
      __builtin_prefetch(W + wrow + k0 + 32 + scol, 0, 0);
    }
    // stage A = X tile (128x32), convert f32->bf16 if needed
    if constexpr (IN_BF16) {
      // bf16 input: pure copy -> async DMA path (32 bf16 = 64 bytes)
      if ((tid & 1) == 0)
        async_copy64(&As[srow][0], Xb + xrow + k0);
    } else {
      const float* xp = Xf + xrow + k0 + scol;
      #pragma unroll
      for (int j = 0; j < 16; ++j) As[srow][scol + j] = f2bf(xp[j]);
    }
    // stage B = W rows n0..n0+127, cols k0..k0+31 (f32 -> bf16)
    {
      const float* wp = W + wrow + k0 + scol;
      #pragma unroll
      for (int j = 0; j < 16; ++j) Bs[srow][scol + j] = f2bf(wp[j]);
    }
    if constexpr (IN_BF16) wait_async0();
    __syncthreads();

    // A fragment: lane r holds row M=r, K chunks {lh*8..} and {16+lh*8..}
    v16bf a[2];
    #pragma unroll
    for (int mt = 0; mt < 2; ++mt) {
      const __bf16* p = &As[mBase + mt * 16 + r][lh * 8];
      a[mt] = cat8(*(const v8bf*)p, *(const v8bf*)(p + 16));
    }
    // B fragment: lane r holds column N=r, K chunks {lh*8..} and {16+lh*8..}
    #pragma unroll
    for (int nt = 0; nt < 4; ++nt) {
      const __bf16* p = &Bs[nBase + nt * 16 + r][lh * 8];
      v16bf b = cat8(*(const v8bf*)p, *(const v8bf*)(p + 16));
      #pragma unroll
      for (int mt = 0; mt < 2; ++mt)
        c[mt][nt] = wmma_bf16(a[mt], b, c[mt][nt]);
    }
  }

  // epilogue: C/D layout -> element j of lane = (M = j + 8*lh, N = r)
  #pragma unroll
  for (int mt = 0; mt < 2; ++mt) {
    #pragma unroll
    for (int nt = 0; nt < 4; ++nt) {
      const int gn = n0 + nBase + nt * 16 + r;
      const float bb = bias[gn];
      #pragma unroll
      for (int j = 0; j < 8; ++j) {
        const int gm = m0 + mBase + mt * 16 + lh * 8 + j;
        const float v = c[mt][nt][j] + bb;
        if constexpr (OUT_TRANSV) {
          // row gm = bn*1024 + key ; col gn = h*64 + hd
          const int bnI = gm >> 10;
          const int key = gm & (CK - 1);
          const int hh  = gn >> 6;
          const int hd  = gn & (CHD - 1);
          ((__bf16*)Yv)[(((size_t)bnI * CH + hh) * CHD + hd) * CK + key] =
              f2bf(v);   // j-consecutive -> key-contiguous stores
        } else if constexpr (OUT_BF16) {
          ((__bf16*)Yv)[(size_t)gm * CD + gn] = f2bf(v);
        } else {
          ((float*)Yv)[(size_t)gm * CD + gn] = v;
        }
      }
    }
  }
}

// ---------------------------------------------------------------------------
// Fused attention per (b, n, h): flash-style over 8 key blocks of 128.
// 8 waves; wave w owns q rows [w*16, w*16+16).
// K and V^T tiles double-buffered via async DMA: block i+1 streams into LDS
// while block i runs on the WMMA pipes.
// ---------------------------------------------------------------------------
__global__ __launch_bounds__(256) void attn_fused(
    const __bf16* __restrict__ qb, const __bf16* __restrict__ kbuf,
    const __bf16* __restrict__ vbt, const float* __restrict__ qpos,
    const float* __restrict__ kpos, const unsigned char* __restrict__ kmask,
    __bf16* __restrict__ ob)
{
  __shared__ __bf16 Qs[CQ][CHD + 8];        // q tile, row-major [q][hd]
  __shared__ __bf16 Ks[2][128][CHD + 8];    // key blocks, row-major [key][hd]
  __shared__ __bf16 VTs[2][CHD][136];       // v blocks, transposed [hd][key]
  __shared__ __bf16 Ps[8][16][136];         // per-wave P re-layout buffer
  __shared__ float QPx[CQ], QPy[CQ], QPz[CQ];
  __shared__ float KPx[2][128], KPy[2][128], KPz[2][128], KMv[2][128];

  const int idx = blockIdx.x;
  const int h = idx & (CH - 1);
  const int n = (idx / CH) & (CN - 1);
  const int b = idx / (CH * CN);
  const int bn = b * CN + n;

  const int tid  = threadIdx.x;
  const int lane = tid & 31;
  const int w    = tid >> 5;
  const int lh   = lane >> 4;
  const int r    = lane & 15;

  const int srow = tid >> 1;             // K staging row (0..127), 2 thr/row
  const bool slo = (tid & 1) == 0;       // even thread stages the K row
  const int vrow = tid >> 2;             // V^T staging row hd (0..63)
  const int vch  = (tid & 3) * 32;       // V^T key chunk (32 keys = 64B)

  const size_t vhead = ((size_t)bn * CH + h) * CHD;  // base row of V^T head

  auto stage_kv = [&](int kb0, int bsel) {
    if (slo)
      async_copy64(&Ks[bsel][srow][0],
                   kbuf + ((size_t)bn * CK + kb0 + srow) * CD + h * CHD);
    async_copy64(&VTs[bsel][vrow][vch],
                 vbt + (vhead + vrow) * CK + kb0 + vch);
  };
  auto stage_pos = [&](int kb0, int bsel) {
    if (tid < 128) {
      const float* p = kpos + ((size_t)b * CK + kb0 + tid) * 3;
      KPx[bsel][tid] = p[0]; KPy[bsel][tid] = p[1]; KPz[bsel][tid] = p[2];
      KMv[bsel][tid] = kmask[(size_t)b * CK + kb0 + tid] ? 1.0f : 0.0f;
    }
  };

  // prologue: stage Q tile + block 0 K/V^T (all async) + positions
  if (slo)
    async_copy64(&Qs[srow][0], qb + ((size_t)bn * CQ + srow) * CD + h * CHD);
  stage_kv(0, 0);
  stage_pos(0, 0);
  if (tid < CQ) {
    const float* p = qpos + ((size_t)b * CQ + tid) * 3;
    QPx[tid] = p[0]; QPy[tid] = p[1]; QPz[tid] = p[2];
  }
  wait_async0();
  __syncthreads();

  // q fragments (reused across all key blocks): 16x64 as two 16x32
  v16bf aq[2];
  #pragma unroll
  for (int ks = 0; ks < 2; ++ks) {
    const __bf16* p = &Qs[w * 16 + r][ks * 32 + lh * 8];
    aq[ks] = cat8(*(const v8bf*)p, *(const v8bf*)(p + 16));
  }
  float qx[8], qy[8], qz[8];
  #pragma unroll
  for (int j = 0; j < 8; ++j) {
    const int qr = w * 16 + lh * 8 + j;
    qx[j] = QPx[qr]; qy[j] = QPy[qr]; qz[j] = QPz[qr];
  }

  v8f o[4] = {};
  float m_s[8], l_s[8];
  #pragma unroll
  for (int j = 0; j < 8; ++j) { m_s[j] = -1e30f; l_s[j] = 0.0f; }

  for (int blk = 0; blk < CK / 128; ++blk) {
    const int cur = blk & 1;
    // kick off DMA for the next block while this one computes
    if (blk + 1 < CK / 128) {
      stage_kv((blk + 1) * 128, cur ^ 1);
      stage_pos((blk + 1) * 128, cur ^ 1);
    }

    // S = q @ k^T : 16 rows x 128 keys (8 N-subtiles, K=64 in 2 steps)
    v8f s[8];
    #pragma unroll
    for (int nt = 0; nt < 8; ++nt) {
      v8f acc = {};
      #pragma unroll
      for (int ks = 0; ks < 2; ++ks) {
        const __bf16* p = &Ks[cur][nt * 16 + r][ks * 32 + lh * 8];
        v16bf bk = cat8(*(const v8bf*)p, *(const v8bf*)(p + 16));
        acc = wmma_bf16(aq[ks], bk, acc);
      }
      s[nt] = acc;
    }

    // scale + gaussian positional bias + key mask (in C-layout registers)
    #pragma unroll
    for (int nt = 0; nt < 8; ++nt) {
      const int kl = nt * 16 + r;
      const float kxx = KPx[cur][kl], kyy = KPy[cur][kl], kzz = KPz[cur][kl];
      const float valid = KMv[cur][kl];
      #pragma unroll
      for (int j = 0; j < 8; ++j) {
        const float dx = qx[j] - kxx, dy = qy[j] - kyy, dz = qz[j] - kzz;
        const float pb = -(dx * dx + dy * dy + dz * dz) * CINV2SIG;
        s[nt][j] = (valid != 0.0f) ? (s[nt][j] * CSCALE + pb) : -10000.0f;
      }
    }

    // online softmax: rows live in 16-lane halves -> xor-shuffle reductions
    float sf[8];
    #pragma unroll
    for (int j = 0; j < 8; ++j) {
      float v = s[0][j];
      #pragma unroll
      for (int nt = 1; nt < 8; ++nt) v = fmaxf(v, s[nt][j]);
      v = fmaxf(v, __shfl_xor(v, 1));
      v = fmaxf(v, __shfl_xor(v, 2));
      v = fmaxf(v, __shfl_xor(v, 4));
      v = fmaxf(v, __shfl_xor(v, 8));
      const float mn = fmaxf(m_s[j], v);
      sf[j] = __expf(m_s[j] - mn);
      m_s[j] = mn;
    }
    float rs[8];
    #pragma unroll
    for (int j = 0; j < 8; ++j) rs[j] = 0.0f;
    #pragma unroll
    for (int nt = 0; nt < 8; ++nt) {
      #pragma unroll
      for (int j = 0; j < 8; ++j) {
        const float pv = __expf(s[nt][j] - m_s[j]);
        rs[j] += pv;
        Ps[w][lh * 8 + j][nt * 16 + r] = f2bf(pv);  // C-layout -> row-major
      }
    }
    #pragma unroll
    for (int j = 0; j < 8; ++j) {
      float v = rs[j];
      v += __shfl_xor(v, 1);
      v += __shfl_xor(v, 2);
      v += __shfl_xor(v, 4);
      v += __shfl_xor(v, 8);
      l_s[j] = l_s[j] * sf[j] + v;
      #pragma unroll
      for (int nt2 = 0; nt2 < 4; ++nt2) o[nt2][j] *= sf[j];
    }

    // O += P @ V : 16x128 @ 128x64 (4 K-steps of 32, 4 hd-subtiles)
    #pragma unroll
    for (int ks2 = 0; ks2 < 4; ++ks2) {
      const __bf16* p = &Ps[w][r][ks2 * 32 + lh * 8];
      v16bf ap = cat8(*(const v8bf*)p, *(const v8bf*)(p + 16));
      #pragma unroll
      for (int nt2 = 0; nt2 < 4; ++nt2) {
        const __bf16* pv = &VTs[cur][nt2 * 16 + r][ks2 * 32 + lh * 8];
        v16bf bv = cat8(*(const v8bf*)pv, *(const v8bf*)(pv + 16));
        o[nt2] = wmma_bf16(ap, bv, o[nt2]);
      }
    }

    wait_async0();     // next block's tiles have landed in LDS
    __syncthreads();   // everyone done reading buf[cur]; new tiles visible
  }

  // normalize and store bf16 attention output
  #pragma unroll
  for (int j = 0; j < 8; ++j) {
    const float inv = 1.0f / l_s[j];
    const int qr = w * 16 + lh * 8 + j;
    #pragma unroll
    for (int nt2 = 0; nt2 < 4; ++nt2) {
      ob[((size_t)bn * CQ + qr) * CD + h * CHD + nt2 * 16 + r] =
          f2bf(o[nt2][j] * inv);
    }
  }
}

// ---------------------------------------------------------------------------
extern "C" void kernel_launch(void* const* d_in, const int* in_sizes, int n_in,
                              void* d_out, int out_size, void* d_ws, size_t ws_size,
                              hipStream_t stream) {
  (void)in_sizes; (void)n_in; (void)out_size; (void)ws_size;

  const float* query      = (const float*)d_in[0];
  const float* key_value  = (const float*)d_in[1];
  const float* qpos       = (const float*)d_in[2];
  const float* kpos       = (const float*)d_in[3];
  const unsigned char* km = (const unsigned char*)d_in[4];  // bool array
  const float* Wq = (const float*)d_in[5];
  const float* bq = (const float*)d_in[6];
  const float* Wk = (const float*)d_in[7];
  const float* bk = (const float*)d_in[8];
  const float* Wv = (const float*)d_in[9];
  const float* bv = (const float*)d_in[10];
  const float* Wo = (const float*)d_in[11];
  const float* bo = (const float*)d_in[12];

  const size_t Mq  = (size_t)CB * CN * CQ;   // 8192
  const size_t Mkv = (size_t)CB * CN * CK;   // 65536

  __bf16* qb  = (__bf16*)d_ws;
  __bf16* kb  = qb  + Mq  * CD;
  __bf16* vbt = kb  + Mkv * CD;   // V stored head-transposed [bn][h][hd][key]
  __bf16* ob  = vbt + Mkv * CD;   // total ~151 MB bf16 scratch

  dim3 blk(256);
  gemm_xwt_512<false, true,  false><<<dim3((unsigned)(Mq  / 128) * 4), blk, 0, stream>>>(query, Wq, bq, qb);
  gemm_xwt_512<false, true,  false><<<dim3((unsigned)(Mkv / 128) * 4), blk, 0, stream>>>(key_value, Wk, bk, kb);
  gemm_xwt_512<false, true,  true ><<<dim3((unsigned)(Mkv / 128) * 4), blk, 0, stream>>>(key_value, Wv, bv, vbt);
  attn_fused<<<dim3(CB * CN * CH), blk, 0, stream>>>(qb, kb, vbt, qpos, kpos, km, ob);
  gemm_xwt_512<true,  false, false><<<dim3((unsigned)(Mq / 128) * 4), blk, 0, stream>>>(ob, Wo, bo, (float*)d_out);
}
